// MyLSTM_89575837925760
// MI455X (gfx1250) — matrix operand order repaired and checked by reading, not compiled, not run on hardware
//
#include <hip/hip_runtime.h>

// ---------------------------------------------------------------------------
// LSTM (B=64, S=512, I=512, H=1024) for MI455X / gfx1250.
//   Phase 1 (parallel):  gx = x @ w_ih^T + (b_ih + b_hh)   [f16 WMMA, f32 acc]
//   Phase 2 (serial):    512x fused step: gates = gx_t + h @ w_hh^T ; cell
// All matmuls use V_WMMA_F32_16X16X32_F16. Weights are [N,K] row-major with K
// contiguous, and this code feeds A and B in contiguous k order (16
// consecutive K per lane) -> pure contiguous b128 loads, no transposes.
// ---------------------------------------------------------------------------

typedef __attribute__((ext_vector_type(16))) _Float16 v16h;
typedef __attribute__((ext_vector_type(8)))  _Float16 v8h;
typedef __attribute__((ext_vector_type(8)))  float    v8f;

#define WMMA_F16(a, b, c) \
  __builtin_amdgcn_wmma_f32_16x16x32_f16(false, (a), false, (b), (short)0, (c), false, false)

static constexpr int Bsz = 64;    // batch
static constexpr int Ssz = 512;   // sequence
static constexpr int Isz = 512;   // input dim
static constexpr int Hsz = 1024;  // hidden dim
static constexpr int Gsz = 4 * Hsz;

// Combine two 8-half chunks into one 16-half A/B fragment register group.
__device__ __forceinline__ v16h combine16(v8h lo, v8h hi) {
  return __builtin_shufflevector(lo, hi, 0,1,2,3,4,5,6,7,8,9,10,11,12,13,14,15);
}

// A-fragment (16x32 f16, MxK): lane half h = lane>>4, row m = lane&15.
//   elements 0..7  = K [k0 + 16*h,     +8)
//   elements 8..15 = K [k0 + 16*h + 8, +8)
__device__ __forceinline__ v16h load_a_frag(const _Float16* __restrict__ row, int k0, int half) {
  v8h lo = *(const v8h*)(row + k0 + 16 * half);
  v8h hi = *(const v8h*)(row + k0 + 16 * half + 8);
  return combine16(lo, hi);
}

__device__ __forceinline__ float sigmoidf_fast(float x) {
  return 1.0f / (1.0f + __expf(-x));
}

// ---------------------------------------------------------------------------
__global__ void f32_to_f16_kernel(const float* __restrict__ src,
                                  _Float16* __restrict__ dst, int n) {
  int i = blockIdx.x * blockDim.x + threadIdx.x;
  int stride = gridDim.x * blockDim.x;
  for (; i < n; i += stride) dst[i] = (_Float16)src[i];
}

__global__ void init_state_kernel(_Float16* __restrict__ h0,
                                  float* __restrict__ c, int n) {
  int i = blockIdx.x * blockDim.x + threadIdx.x;
  if (i < n) { h0[i] = (_Float16)0.0f; c[i] = 0.0f; }
}

// ---------------------------------------------------------------------------
// gx[s*B+b][g] = sum_k x[b][s][k] * w_ih[g][k] + b_ih[g] + b_hh[g]
// M = S*B = 32768, N = 4H = 4096, K = I = 512.
// Per wave: 32(M) x 64(N) tile = 2x4 accumulators; K-loop in chunks of 32.
__global__ __launch_bounds__(256)
void gx_gemm_kernel(const _Float16* __restrict__ xh,
                    const _Float16* __restrict__ wih,
                    const float* __restrict__ b_ih,
                    const float* __restrict__ b_hh,
                    float* __restrict__ gx) {
  const int lane = threadIdx.x & 31;
  const int wave = blockIdx.x * 8 + (threadIdx.x >> 5);   // 65536 waves
  const int nt   = wave & 63;                             // 64 N-tiles of 64
  const int mt   = wave >> 6;                             // 1024 M-tiles of 32
  const int half = lane >> 4;
  const int lidx = lane & 15;

  const int n0 = nt * 64;
  const int m0 = mt * 32;

  // Global M row r = s*B + b  ->  s = r>>6, b = r&63; x is [B,S,I] row-major.
  const int r0 = m0 + lidx;
  const int r1 = m0 + 16 + lidx;
  const _Float16* arow0 = xh + (size_t)((r0 & 63) * Ssz + (r0 >> 6)) * Isz;
  const _Float16* arow1 = xh + (size_t)((r1 & 63) * Ssz + (r1 >> 6)) * Isz;

  v8f acc[2][4];
#pragma unroll
  for (int u = 0; u < 2; ++u)
#pragma unroll
    for (int t = 0; t < 4; ++t) acc[u][t] = (v8f)0.0f;

  for (int k = 0; k < Isz; k += 32) {
    if (k + 32 < Isz) {
      __builtin_prefetch(arow0 + k + 32, 0, 1);
      __builtin_prefetch(arow1 + k + 32, 0, 1);
    }
    v16h a0 = load_a_frag(arow0, k, half);
    v16h a1 = load_a_frag(arow1, k, half);
#pragma unroll
    for (int t = 0; t < 4; ++t) {
      // B-fragment (32x16): lane column n = lidx, 16 consecutive K per lane.
      const _Float16* brow = wih + (size_t)(n0 + t * 16 + lidx) * Isz + k + 16 * half;
      v16h b = *(const v16h*)brow;
      acc[0][t] = WMMA_F16(a0, b, acc[0][t]);
      acc[1][t] = WMMA_F16(a1, b, acc[1][t]);
    }
  }

  // C layout: VGPR j, lane -> (M = 8*half + j, N = lidx) within 16x16 tile.
#pragma unroll
  for (int t = 0; t < 4; ++t) {
    const int n = n0 + t * 16 + lidx;
    const float bias = b_ih[n] + b_hh[n];
#pragma unroll
    for (int u = 0; u < 2; ++u) {
      const int rbase = m0 + u * 16 + half * 8;
      float* outp = gx + (size_t)rbase * Gsz + n;
#pragma unroll
      for (int j = 0; j < 8; ++j)
        outp[(size_t)j * Gsz] = acc[u][t][j] + bias;
    }
  }
}

// ---------------------------------------------------------------------------
// One recurrence step: gates = gx[s] + h_in @ w_hh^T, then fused LSTM cell.
// 256 waves: wave (mt, cb) owns batch rows [mt*16,+16) and hidden columns
// [cb*16,+16) for ALL FOUR gates (g-offsets 0,H,2H,3H) -> elementwise cell is
// fused in registers; c updated in place (exclusive ownership), h ping-pongs.
__global__ __launch_bounds__(128)
void lstm_step_kernel(const float* __restrict__ gx,
                      const _Float16* __restrict__ whh,
                      const _Float16* __restrict__ hin,
                      _Float16* __restrict__ hout,
                      float* __restrict__ c,
                      float* __restrict__ hseq,
                      int s) {
  const int lane = threadIdx.x & 31;
  const int wave = blockIdx.x * 4 + (threadIdx.x >> 5);   // 256 waves
  const int cb   = wave & 63;                             // hidden col block
  const int mt   = wave >> 6;                             // batch tile (0..3)
  const int half = lane >> 4;
  const int lidx = lane & 15;

  const int m = mt * 16 + lidx;                 // batch row for A fragment
  const _Float16* arow = hin + (size_t)m * Hsz;

  v8f acc[4];
#pragma unroll
  for (int g = 0; g < 4; ++g) acc[g] = (v8f)0.0f;

  for (int k = 0; k < Hsz; k += 32) {
    if (k + 32 < Hsz) __builtin_prefetch(arow + k + 32, 0, 1);
    v16h a = load_a_frag(arow, k, half);
#pragma unroll
    for (int g = 0; g < 4; ++g) {
      const _Float16* brow =
          whh + (size_t)(g * Hsz + cb * 16 + lidx) * Hsz + k + 16 * half;
      v16h b = *(const v16h*)brow;
      acc[g] = WMMA_F16(a, b, acc[g]);
    }
  }

  const int n = cb * 16 + lidx;
#pragma unroll
  for (int j = 0; j < 8; ++j) {
    const int bb = mt * 16 + half * 8 + j;
    const float* gxr = gx + ((size_t)s * Bsz + bb) * Gsz;
    float iv = sigmoidf_fast(acc[0][j] + gxr[n]);
    float fv = sigmoidf_fast(acc[1][j] + gxr[Hsz + n]);
    float gv = tanhf(acc[2][j] + gxr[2 * Hsz + n]);
    float ov = sigmoidf_fast(acc[3][j] + gxr[3 * Hsz + n]);
    const size_t cidx = (size_t)bb * Hsz + n;
    const float cn = fv * c[cidx] + iv * gv;
    const float hn = ov * tanhf(cn);
    c[cidx] = cn;
    hseq[((size_t)s * Bsz + bb) * Hsz + n] = hn;
    hout[cidx] = (_Float16)hn;
  }
}

// ---------------------------------------------------------------------------
__global__ void tail_kernel(const float* __restrict__ h_last,
                            const float* __restrict__ c,
                            float* __restrict__ out_tail, int n) {
  int i = blockIdx.x * blockDim.x + threadIdx.x;
  if (i < n) {
    out_tail[i]     = h_last[i];  // h_f
    out_tail[n + i] = c[i];       // c_f
  }
}

// ---------------------------------------------------------------------------
extern "C" void kernel_launch(void* const* d_in, const int* in_sizes, int n_in,
                              void* d_out, int out_size, void* d_ws, size_t ws_size,
                              hipStream_t stream) {
  (void)in_sizes; (void)n_in; (void)out_size; (void)ws_size;
  const float* x    = (const float*)d_in[0];  // [B,S,I]
  const float* w_ih = (const float*)d_in[1];  // [4H,I]
  const float* w_hh = (const float*)d_in[2];  // [4H,H]
  const float* b_ih = (const float*)d_in[3];  // [4H]
  const float* b_hh = (const float*)d_in[4];  // [4H]
  float* out = (float*)d_out;                 // hseq [S,B,H] ++ h_f ++ c_f

  // Workspace carve-up (all chunk sizes are multiples of 256B).
  char* ws = (char*)d_ws;
  _Float16* xh    = (_Float16*)ws; ws += (size_t)Bsz * Ssz * Isz * sizeof(_Float16);
  _Float16* wih16 = (_Float16*)ws; ws += (size_t)Gsz * Isz * sizeof(_Float16);
  _Float16* whh16 = (_Float16*)ws; ws += (size_t)Gsz * Hsz * sizeof(_Float16);
  float*    gx    = (float*)ws;    ws += (size_t)Ssz * Bsz * Gsz * sizeof(float);
  _Float16* hb0   = (_Float16*)ws; ws += (size_t)Bsz * Hsz * sizeof(_Float16);
  _Float16* hb1   = (_Float16*)ws; ws += (size_t)Bsz * Hsz * sizeof(_Float16);
  float*    cbuf  = (float*)ws;

  // Precision setup: f16 copies for WMMA operands.
  f32_to_f16_kernel<<<4096, 256, 0, stream>>>(x,    xh,    Bsz * Ssz * Isz);
  f32_to_f16_kernel<<<2048, 256, 0, stream>>>(w_ih, wih16, Gsz * Isz);
  f32_to_f16_kernel<<<4096, 256, 0, stream>>>(w_hh, whh16, Gsz * Hsz);
  init_state_kernel<<<(Bsz * Hsz + 255) / 256, 256, 0, stream>>>(hb0, cbuf, Bsz * Hsz);

  // Phase 1: big parallel GEMM for the input contribution (+fused bias).
  gx_gemm_kernel<<<8192, 256, 0, stream>>>(xh, wih16, b_ih, b_hh, gx);

  // Phase 2: sequential recurrence; w_hh (8MB f16) lives in the 192MB L2.
  _Float16* hp[2] = {hb0, hb1};
  for (int s = 0; s < Ssz; ++s) {
    lstm_step_kernel<<<64, 128, 0, stream>>>(gx, whh16, hp[s & 1], hp[(s + 1) & 1],
                                             cbuf, out, s);
  }

  // Final (h_f, c_f) tuple after the hidden sequence.
  tail_kernel<<<(Bsz * Hsz + 255) / 256, 256, 0, stream>>>(
      out + (size_t)(Ssz - 1) * Bsz * Hsz, cbuf,
      out + (size_t)Ssz * Bsz * Hsz, Bsz * Hsz);
}
